// MultiHeadAttention_24361054503516
// MI455X (gfx1250) — matrix operand order repaired and checked by reading, not compiled, not run on hardware
//
#include <hip/hip_runtime.h>
#include <hip/hip_bf16.h>
#include <stdint.h>

typedef __attribute__((ext_vector_type(16))) __bf16 bhalf16;
typedef __attribute__((ext_vector_type(8)))  float  v8f;

// ---------------------------------------------------------------- helpers

__device__ __forceinline__ unsigned short f2bf(float x) {
    union { float f; unsigned u; } t; t.f = x;
    unsigned r = t.u + 0x7fffu + ((t.u >> 16) & 1u);   // round-to-nearest-even
    return (unsigned short)(r >> 16);
}

union FragU {
    bhalf16 v;
    unsigned short u[16];
    uint4 q[2];
};

// 16 contiguous bf16, 32B-aligned source (unused; B fragments here use the 2x8 pattern below)
__device__ __forceinline__ bhalf16 load_frag_contig(const unsigned short* p) {
    FragU t;
    t.q[0] = *(const uint4*)(p);
    t.q[1] = *(const uint4*)(p + 8);
    return t.v;
}

// two 8-element contiguous runs, each 16B-aligned (A-fragment pattern:
// K = (j>>3)*16 + half*8 + (j&7) -> runs at base+half*8 and base+16+half*8)
__device__ __forceinline__ bhalf16 load_frag_2x8(const unsigned short* p0,
                                                 const unsigned short* p1) {
    FragU t;
    t.q[0] = *(const uint4*)(p0);
    t.q[1] = *(const uint4*)(p1);
    return t.v;
}

// ---------------------------------------------------------------- converts

__global__ void k_conv_bf16(const float* __restrict__ in,
                            unsigned short* __restrict__ out, int n) {
    int i = blockIdx.x * 256 + threadIdx.x;
    if (i < n) out[i] = f2bf(in[i]);
}

// in: [H,C,D] = [16,1024,64] fp32  ->  out: [H,D,C] bf16 (transposed per head)
__global__ void k_conv_wT(const float* __restrict__ in,
                          unsigned short* __restrict__ out) {
    int i = blockIdx.x * 256 + threadIdx.x;     // over H*D*C = 1M, out index
    int c  = i & 1023;
    int hd = i >> 10;
    int d  = hd & 63;
    int h  = hd >> 6;
    out[i] = f2bf(in[((size_t)(h * 1024 + c)) * 64 + d]);
}

// ---------------------------------------------------------------- QKV GEMM
// rows r = b*T + t over 4096; per head: [4096,1024] x [1024,64] -> [4096,64]
// grid (32, H=16, 3); block 128 (4 waves); wave computes 32x64 (2 A frags
// amortize the 4 weight B-fragments -> 8 WMMA per 6 fragment loads).

__global__ __launch_bounds__(128)
void k_qkv(const unsigned short* __restrict__ xb,
           const unsigned short* __restrict__ wqT,
           const unsigned short* __restrict__ wkT,
           const unsigned short* __restrict__ wvT,
           unsigned short* __restrict__ qo,     // [B,H,T,D]
           unsigned short* __restrict__ ko,     // [B,H,T,D]
           unsigned short* __restrict__ vTo) {  // [B,H,D,T]
    const int lane = threadIdx.x & 31;
    const int wave = threadIdx.x >> 5;
    const int half = lane >> 4;
    const int nl   = lane & 15;
    const int head = blockIdx.y;
    const int z    = blockIdx.z;
    const unsigned short* wT = (z == 0) ? wqT : (z == 1) ? wkT : wvT;
    const int r0 = blockIdx.x * 128 + wave * 32;

    v8f acc[2][4] = {};
    const unsigned short* arow0 = xb + (size_t)(r0 + nl) * 1024;
    const unsigned short* arow1 = xb + (size_t)(r0 + 16 + nl) * 1024;
    const unsigned short* wbase = wT + (size_t)head * 64 * 1024;

    for (int kb = 0; kb < 1024; kb += 32) {
        bhalf16 a0 = load_frag_2x8(arow0 + kb + half * 8,
                                   arow0 + kb + 16 + half * 8);
        bhalf16 a1 = load_frag_2x8(arow1 + kb + half * 8,
                                   arow1 + kb + 16 + half * 8);
#pragma unroll
        for (int nt = 0; nt < 4; ++nt) {
            const unsigned short* bp =
                wbase + (size_t)(nt * 16 + nl) * 1024 + kb + half * 8;
            bhalf16 bfr = load_frag_2x8(bp, bp + 16);
            acc[0][nt] = __builtin_amdgcn_wmma_f32_16x16x32_bf16(
                false, a0, false, bfr, (short)0, acc[0][nt], false, false);
            acc[1][nt] = __builtin_amdgcn_wmma_f32_16x16x32_bf16(
                false, a1, false, bfr, (short)0, acc[1][nt], false, false);
        }
    }

#pragma unroll
    for (int mt = 0; mt < 2; ++mt) {
#pragma unroll
        for (int nt = 0; nt < 4; ++nt) {
#pragma unroll
            for (int i = 0; i < 8; ++i) {
                int r = r0 + mt * 16 + half * 8 + i;
                int b = r >> 10, t = r & 1023;
                int d = nt * 16 + nl;
                unsigned short val = f2bf(acc[mt][nt][i]);
                size_t bh = (size_t)(b * 16 + head);
                if (z == 0)      qo [(bh * 1024 + t) * 64 + d]  = val;
                else if (z == 1) ko [(bh * 1024 + t) * 64 + d]  = val;
                else             vTo[(bh * 64 + d) * 1024 + t]  = val;
            }
        }
    }
}

// ---------------------------------------------------------------- attention
// grid (T/64=16, B*H=64); block 128; wave owns 16 query rows, flash softmax.

__global__ __launch_bounds__(128)
void k_attn(const unsigned short* __restrict__ qg,   // [B,H,T,D]
            const unsigned short* __restrict__ kg,   // [B,H,T,D]
            const unsigned short* __restrict__ vTg,  // [B,H,D,T]
            unsigned short* __restrict__ cat) {      // [B,T,H*D]
    __shared__ __align__(16) unsigned short Plds[4][16][32];

    const int lane = threadIdx.x & 31;
    const int wave = threadIdx.x >> 5;
    const int half = lane >> 4;
    const int nl   = lane & 15;
    const int bh   = blockIdx.y;           // b*16 + h
    const int b    = bh >> 4, h = bh & 15;
    const int q0   = blockIdx.x * 64 + wave * 16;
    const float scale = 0.125f;            // 1/sqrt(64)
    const float NEG = -3.0e38f;

    const unsigned short* qp = qg  + (size_t)bh * 1024 * 64;
    const unsigned short* kp = kg  + (size_t)bh * 1024 * 64;
    const unsigned short* vp = vTg + (size_t)bh * 64 * 1024;

    // Q A-fragments (contraction over D=64 -> two 16x32 fragments), kept live
    bhalf16 qa[2];
    {
        const unsigned short* qrow = qp + (size_t)(q0 + nl) * 64;
        qa[0] = load_frag_2x8(qrow + half * 8,      qrow + 16 + half * 8);
        qa[1] = load_frag_2x8(qrow + 32 + half * 8, qrow + 48 + half * 8);
    }

    float mi[8], li[8];
    v8f o[4] = {};
#pragma unroll
    for (int i = 0; i < 8; ++i) { mi[i] = NEG; li[i] = 0.0f; }

    const int send = q0 + 16;              // keys needed: s <= q0+15
    for (int sc = 0; sc < send; sc += 32) {
        // ---- scores: two 16x16 C-tiles covering keys [sc, sc+32)
        v8f s[2];
#pragma unroll
        for (int st = 0; st < 2; ++st) {
            v8f a = {};
#pragma unroll
            for (int f = 0; f < 2; ++f) {
                const unsigned short* bp =
                    kp + (size_t)(sc + st * 16 + nl) * 64 + f * 32 + half * 8;
                bhalf16 kfr = load_frag_2x8(bp, bp + 16);
                a = __builtin_amdgcn_wmma_f32_16x16x32_bf16(
                    false, qa[f], false, kfr, (short)0, a, false, false);
            }
            s[st] = a;
        }

        // ---- causal mask + scale, running row max
        float rmax[8];
#pragma unroll
        for (int i = 0; i < 8; ++i) rmax[i] = mi[i];
#pragma unroll
        for (int st = 0; st < 2; ++st) {
#pragma unroll
            for (int i = 0; i < 8; ++i) {
                int t_g = q0 + half * 8 + i;
                int s_g = sc + st * 16 + nl;
                float v = s[st][i] * scale;
                v = (s_g <= t_g) ? v : NEG;
                s[st][i] = v;
                rmax[i] = fmaxf(rmax[i], v);
            }
        }
#pragma unroll
        for (int i = 0; i < 8; ++i) {
#pragma unroll
            for (int off = 1; off < 16; off <<= 1)
                rmax[i] = fmaxf(rmax[i], __shfl_xor(rmax[i], off, 32));
        }

        // ---- rescale state
        float alpha[8];
#pragma unroll
        for (int i = 0; i < 8; ++i) {
            alpha[i] = __expf(mi[i] - rmax[i]);
            mi[i] = rmax[i];
            li[i] *= alpha[i];
        }

        // ---- exponentiate, row sums, stage P (bf16) into LDS
        float rsum[8] = {0.f,0.f,0.f,0.f,0.f,0.f,0.f,0.f};
#pragma unroll
        for (int st = 0; st < 2; ++st) {
#pragma unroll
            for (int i = 0; i < 8; ++i) {
                float p = __expf(s[st][i] - mi[i]);
                rsum[i] += p;
                Plds[wave][half * 8 + i][st * 16 + nl] = f2bf(p);
            }
        }
#pragma unroll
        for (int i = 0; i < 8; ++i) {
#pragma unroll
            for (int off = 1; off < 16; off <<= 1)
                rsum[i] += __shfl_xor(rsum[i], off, 32);
            li[i] += rsum[i];
        }

        // ---- O = O*alpha + P @ V    (same-wave LDS ops are in-order)
#pragma unroll
        for (int dt = 0; dt < 4; ++dt)
#pragma unroll
            for (int i = 0; i < 8; ++i) o[dt][i] *= alpha[i];

        const unsigned short* prow = &Plds[wave][nl][0];
        bhalf16 pa = load_frag_2x8(prow + half * 8, prow + 16 + half * 8);
#pragma unroll
        for (int dt = 0; dt < 4; ++dt) {
            const unsigned short* bp =
                vp + (size_t)(dt * 16 + nl) * 1024 + sc + half * 8;
            bhalf16 vfr = load_frag_2x8(bp, bp + 16);
            o[dt] = __builtin_amdgcn_wmma_f32_16x16x32_bf16(
                false, pa, false, vfr, (short)0, o[dt], false, false);
        }
    }

    // ---- normalize + write concat buffer [B,T,H*D] as bf16
#pragma unroll
    for (int dt = 0; dt < 4; ++dt) {
#pragma unroll
        for (int i = 0; i < 8; ++i) {
            int t = q0 + half * 8 + i;
            float vout = o[dt][i] / li[i];
            cat[((size_t)b * 1024 + t) * 1024 + h * 64 + dt * 16 + nl] =
                f2bf(vout);
        }
    }
}

// ---------------------------------------------------------------- out proj
// [4096,1024] x Wo^T (B[k=i][n=c] = Wo[c,i], contiguous) + bias -> f32 out
// grid (32, 16); wave computes 32x64 like k_qkv.

__global__ __launch_bounds__(128)
void k_oproj(const unsigned short* __restrict__ cat,
             const unsigned short* __restrict__ wo,
             const float* __restrict__ bo,
             float* __restrict__ out) {
    const int lane = threadIdx.x & 31;
    const int wave = threadIdx.x >> 5;
    const int half = lane >> 4;
    const int nl   = lane & 15;
    const int r0 = blockIdx.x * 128 + wave * 32;
    const int c0 = blockIdx.y * 64;

    v8f acc[2][4] = {};
    const unsigned short* arow0 = cat + (size_t)(r0 + nl) * 1024;
    const unsigned short* arow1 = cat + (size_t)(r0 + 16 + nl) * 1024;

    for (int kb = 0; kb < 1024; kb += 32) {
        bhalf16 a0 = load_frag_2x8(arow0 + kb + half * 8,
                                   arow0 + kb + 16 + half * 8);
        bhalf16 a1 = load_frag_2x8(arow1 + kb + half * 8,
                                   arow1 + kb + 16 + half * 8);
#pragma unroll
        for (int nt = 0; nt < 4; ++nt) {
            const unsigned short* bp =
                wo + (size_t)(c0 + nt * 16 + nl) * 1024 + kb + half * 8;
            bhalf16 bfr = load_frag_2x8(bp, bp + 16);
            acc[0][nt] = __builtin_amdgcn_wmma_f32_16x16x32_bf16(
                false, a0, false, bfr, (short)0, acc[0][nt], false, false);
            acc[1][nt] = __builtin_amdgcn_wmma_f32_16x16x32_bf16(
                false, a1, false, bfr, (short)0, acc[1][nt], false, false);
        }
    }

#pragma unroll
    for (int mt = 0; mt < 2; ++mt) {
#pragma unroll
        for (int nt = 0; nt < 4; ++nt) {
            int cc = c0 + nt * 16 + nl;
            float bias = bo[cc];
#pragma unroll
            for (int i = 0; i < 8; ++i) {
                int r = r0 + mt * 16 + half * 8 + i;
                out[(size_t)r * 1024 + cc] = acc[mt][nt][i] + bias;
            }
        }
    }
}

// ---------------------------------------------------------------- launch

extern "C" void kernel_launch(void* const* d_in, const int* in_sizes, int n_in,
                              void* d_out, int out_size, void* d_ws, size_t ws_size,
                              hipStream_t stream) {
    const float* x  = (const float*)d_in[0];
    const float* Wq = (const float*)d_in[1];
    const float* Wk = (const float*)d_in[2];
    const float* Wv = (const float*)d_in[3];
    const float* Wo = (const float*)d_in[4];
    const float* bo = (const float*)d_in[5];

    // workspace layout (bf16 elements)
    const size_t NX  = 4u * 1024 * 1024;   // x / q / k / vT / cat
    const size_t NW  = 1024u * 1024;       // per-head weights, Wo
    unsigned short* xb  = (unsigned short*)d_ws;
    unsigned short* wqT = xb  + NX;
    unsigned short* wkT = wqT + NW;
    unsigned short* wvT = wkT + NW;
    unsigned short* wob = wvT + NW;
    unsigned short* qb  = wob + NW;
    unsigned short* kb  = qb  + NX;
    unsigned short* vT  = kb  + NX;
    unsigned short* cat = vT  + NX;        // total 24M elems = 48 MB

    k_conv_bf16<<<(int)(NX / 256), 256, 0, stream>>>(x, xb, (int)NX);
    k_conv_wT  <<<(int)(NW / 256), 256, 0, stream>>>(Wq, wqT);
    k_conv_wT  <<<(int)(NW / 256), 256, 0, stream>>>(Wk, wkT);
    k_conv_wT  <<<(int)(NW / 256), 256, 0, stream>>>(Wv, wvT);
    k_conv_bf16<<<(int)(NW / 256), 256, 0, stream>>>(Wo, wob, (int)NW);

    k_qkv <<<dim3(32, 16, 3), 128, 0, stream>>>(xb, wqT, wkT, wvT, qb, kb, vT);
    k_attn<<<dim3(16, 64),    128, 0, stream>>>(qb, kb, vT, cat);
    k_oproj<<<dim3(32, 16),   128, 0, stream>>>(cat, wob, bo, (float*)d_out);
}
